// LinearRecencyModel_35588099014931
// MI455X (gfx1250) — matrix operand order repaired and checked by reading, not compiled, not run on hardware
//
#include <hip/hip_runtime.h>
#include <hip/hip_bf16.h>

// ---------- sizes ----------
#define Hd    512
#define H2d   1024
#define HALFD 256
#define LSEQ  1024
#define BSZ   64
#define VOC   32000
#define TOK   (BSZ * LSEQ)   // 65536

typedef __attribute__((ext_vector_type(16))) __bf16 v16bf;
typedef __attribute__((ext_vector_type(8)))  float  v8f;
typedef __attribute__((ext_vector_type(2)))  float  v2f;

union BF16Frag { v16bf v; uint4 q[2]; };

// =====================================================================
// Kernel 0: transpose + bf16-convert weights into workspace.
//  w1t: (H2 x H)  row n contiguous in K   (B-matrix for GEMM1)
//  w2t: (H  x H2)                          (B-matrix for GEMM2)
//  pt : (H  x H)  rows 0..255 = sem_w^T, 256..511 = epi_w^T
// =====================================================================
__global__ __launch_bounds__(256) void prep_kernel(
    const float* __restrict__ ff_w1, const float* __restrict__ ff_w2,
    const float* __restrict__ sem_w, const float* __restrict__ epi_w,
    __bf16* __restrict__ w1t, __bf16* __restrict__ w2t, __bf16* __restrict__ pt)
{
  const int S1 = H2d * Hd;   // 524288
  const int S2 = Hd * H2d;   // 524288
  const int S3 = Hd * Hd;    // 262144
  int i = blockIdx.x * 256 + threadIdx.x;
  if (i < S1) {
    int n = i / Hd, k = i % Hd;
    w1t[i] = (__bf16)ff_w1[(size_t)k * H2d + n];
  } else if (i < S1 + S2) {
    int j = i - S1;
    int n = j / H2d, k = j % H2d;
    w2t[j] = (__bf16)ff_w2[(size_t)k * Hd + n];
  } else if (i < S1 + S2 + S3) {
    int j = i - S1 - S2;
    int n = j / Hd, k = j % Hd;
    pt[j] = (__bf16)(n < HALFD ? sem_w[(size_t)k * HALFD + n]
                               : epi_w[(size_t)k * HALFD + (n - HALFD)]);
  }
}

// =====================================================================
// Kernel 1: fused encoder. One workgroup = 16 tokens, 256 threads (8 waves).
//   gather embed -> act1=relu(E@W1+b1) -> ff=act1@W2+b2 -> x=e+ff
//   -> LayerNorm -> [ks|ke] = h @ [sem|epi] + bias  -> ws_k (f32)
// All GEMMs via v_wmma_f32_16x16x32_bf16, f32 accumulate.
// =====================================================================
__global__ __launch_bounds__(256) void encoder_kernel(
    const int*   __restrict__ seq,   const float* __restrict__ embed,
    const __bf16* __restrict__ w1t,  const float* __restrict__ b1,
    const __bf16* __restrict__ w2t,  const float* __restrict__ b2,
    const float* __restrict__ ln_g,  const float* __restrict__ ln_b,
    const __bf16* __restrict__ pt,   const float* __restrict__ sem_b,
    const float* __restrict__ epi_b, float* __restrict__ ws_k)
{
  __shared__ __align__(32) unsigned char smem0[16 * Hd * 2];   // 16KB: E(bf16) then h(bf16)
  __shared__ __align__(32) unsigned char smem1[16 * H2d * 2];  // 32KB: act1(bf16) then x(f32)
  __shared__ float s_mu[16], s_rs[16];
  __shared__ int   s_tok[16];

  __bf16* ebf  = (__bf16*)smem0;
  __bf16* hbf  = (__bf16*)smem0;
  __bf16* act1 = (__bf16*)smem1;
  float*  xf   = (float*)smem1;

  const int tid  = threadIdx.x;
  const int lane = tid & 31, wv = tid >> 5;     // wave32
  const int l15  = lane & 15, hf = lane >> 4;
  const int wg   = blockIdx.x;

  if (tid < 16) s_tok[tid] = seq[wg * 16 + tid];
  __syncthreads();

  // ---- stage gathered embeddings as bf16 (A-matrix for GEMM1) ----
  for (int i = 0; i < 32; ++i) {
    int flat = tid + i * 256;
    int r = flat >> 9, c = flat & (Hd - 1);
    ebf[flat] = (__bf16)embed[(size_t)s_tok[r] * Hd + c];
  }
  __syncthreads();

  // ---- GEMM1: act1 = relu(E @ W1 + b1), 16 x 1024, K=512 ----
  for (int nt = 0; nt < 8; ++nt) {
    int n0 = (wv * 8 + nt) * 16;
    v8f acc = {};
    for (int k0 = 0; k0 < Hd; k0 += 32) {
      BF16Frag a, b;
      a.q[0] = *(const uint4*)&ebf[l15 * Hd + k0 + 8 * hf];
      a.q[1] = *(const uint4*)&ebf[l15 * Hd + k0 + 16 + 8 * hf];
      const uint4* bp = (const uint4*)&w1t[(size_t)(n0 + l15) * Hd + k0 + 8 * hf];
      b.q[0] = bp[0]; b.q[1] = bp[2];
      acc = __builtin_amdgcn_wmma_f32_16x16x32_bf16(false, a.v, false, b.v,
                                                    (short)0, acc, false, false);
    }
    int n = n0 + l15;
    float bias = b1[n];
    for (int r = 0; r < 8; ++r) {
      int m = r + 8 * hf;
      float v = acc[r] + bias;
      act1[m * H2d + n] = (__bf16)(v > 0.f ? v : 0.f);
    }
  }
  __syncthreads();

  // ---- GEMM2: ff = act1 @ W2, 16 x 512, K=1024 (accs held in regs) ----
  v8f acc2[4];
  for (int nt = 0; nt < 4; ++nt) {
    int n0 = (wv * 4 + nt) * 16;
    v8f acc = {};
    for (int k0 = 0; k0 < H2d; k0 += 32) {
      BF16Frag a, b;
      a.q[0] = *(const uint4*)&act1[l15 * H2d + k0 + 8 * hf];
      a.q[1] = *(const uint4*)&act1[l15 * H2d + k0 + 16 + 8 * hf];
      const uint4* bp = (const uint4*)&w2t[(size_t)(n0 + l15) * H2d + k0 + 8 * hf];
      b.q[0] = bp[0]; b.q[1] = bp[2];
      acc = __builtin_amdgcn_wmma_f32_16x16x32_bf16(false, a.v, false, b.v,
                                                    (short)0, acc, false, false);
    }
    acc2[nt] = acc;
  }
  __syncthreads();   // act1 reads done; region becomes x (f32)

  // ---- residual: x = e + ff + b2 (full f32 for exact LN stats) ----
  for (int nt = 0; nt < 4; ++nt) {
    int n = (wv * 4 + nt) * 16 + l15;
    for (int r = 0; r < 8; ++r) {
      int m = r + 8 * hf;
      xf[m * Hd + n] = acc2[nt][r] + b2[n] + embed[(size_t)s_tok[m] * Hd + n];
    }
  }
  __syncthreads();

  // ---- LayerNorm stats: wave wv reduces rows 2wv, 2wv+1 ----
  for (int rr = 0; rr < 2; ++rr) {
    int row = wv * 2 + rr;
    float s = 0.f, q = 0.f;
    for (int i = 0; i < Hd / 32; ++i) {
      float v = xf[row * Hd + lane + i * 32];
      s += v; q += v * v;
    }
    for (int off = 16; off > 0; off >>= 1) {
      s += __shfl_xor(s, off, 32);
      q += __shfl_xor(q, off, 32);
    }
    if (lane == 0) {
      float mu  = s / Hd;
      float var = q / Hd - mu * mu;
      s_mu[row] = mu;
      s_rs[row] = rsqrtf(var + 1e-5f);
    }
  }
  __syncthreads();

  // ---- normalize -> h (bf16) in smem0 ----
  for (int i = 0; i < 32; ++i) {
    int flat = tid + i * 256;
    int r = flat >> 9, c = flat & (Hd - 1);
    float v = (xf[flat] - s_mu[r]) * s_rs[r] * ln_g[c] + ln_b[c];
    hbf[flat] = (__bf16)v;
  }
  __syncthreads();

  // ---- GEMM3: [ks|ke] = h @ PT + bias -> ws_k f32 ----
  for (int nt = 0; nt < 4; ++nt) {
    int n0 = (wv * 4 + nt) * 16;
    v8f acc = {};
    for (int k0 = 0; k0 < Hd; k0 += 32) {
      BF16Frag a, b;
      a.q[0] = *(const uint4*)&hbf[l15 * Hd + k0 + 8 * hf];
      a.q[1] = *(const uint4*)&hbf[l15 * Hd + k0 + 16 + 8 * hf];
      const uint4* bp = (const uint4*)&pt[(size_t)(n0 + l15) * Hd + k0 + 8 * hf];
      b.q[0] = bp[0]; b.q[1] = bp[2];
      acc = __builtin_amdgcn_wmma_f32_16x16x32_bf16(false, a.v, false, b.v,
                                                    (short)0, acc, false, false);
    }
    int n = n0 + l15;
    float bias = (n < HALFD) ? sem_b[n] : epi_b[n - HALFD];
    for (int r = 0; r < 8; ++r) {
      int m = r + 8 * hf;
      size_t t = (size_t)wg * 16 + m;           // global token = b*L + l
      ws_k[t * 512 + n] = acc[r] + bias;
    }
  }
}

// =====================================================================
// Kernel 2: delta-rule scan. One workgroup per (batch, matrix) pair.
// Thread tid owns state row M[tid][0..255] ENTIRELY IN VGPRS (~256 regs,
// uses the gfx1250 VGPR-MSB mechanism). Per step:
//   - kv broadcast vector staged in 1KB static LDS (b128 reads)
//   - ||k||^2 via shfl-xor wave reduce + 8-way LDS combine (deterministic)
//   - vp dot + rank-1 update are pure VALU FMA (co-executes with LDS reads)
//   - next timestep's key prefetched (global_prefetch_b8) to hide latency
//     along the 1023-step serial dependency chain.
// =====================================================================
__global__ __launch_bounds__(256) void scan_kernel(
    const float* __restrict__ ws_k, float* __restrict__ ws_c)
{
  __shared__ __align__(16) float kv[HALFD];
  __shared__ float spart[8];
  const int tid   = threadIdx.x;
  const int lane  = tid & 31, wv = tid >> 5;
  const int b     = blockIdx.x >> 1;
  const int which = blockIdx.x & 1; // 0 = semantic, 1 = episodic
  const float* kbase = ws_k + (size_t)b * LSEQ * 512 + which * HALFD;

  float row[HALFD];
#pragma unroll
  for (int j = 0; j < HALFD; ++j) row[j] = 0.f;

  for (int t = 0; t < LSEQ - 1; ++t) {
    __syncthreads();                              // prior step's kv/spart reads done
    float kt = kbase[(size_t)t * 512 + tid];
    __builtin_prefetch(kbase + (size_t)(t + 1) * 512 + tid, 0, 1);
    kv[tid] = kt;
    // deterministic ||k||^2: wave tree-reduce, then fixed-order 8-way sum
    float p = kt * kt;
    for (int off = 16; off > 0; off >>= 1) p += __shfl_xor(p, off, 32);
    if (lane == 0) spart[wv] = p;
    __syncthreads();
    float s = 1e-6f;                              // DELTA_EPS
#pragma unroll
    for (int w = 0; w < 8; ++w) s += spart[w];

    float vp = 0.f;
#pragma unroll
    for (int j = 0; j < HALFD; j += 4) {
      float4 kq = *(const float4*)&kv[j];
      vp += row[j]     * kq.x;
      vp += row[j + 1] * kq.y;
      vp += row[j + 2] * kq.z;
      vp += row[j + 3] * kq.w;
    }
    float dv   = kt - vp / s;
    float coef = which ? (float)(t + 1) * (1.f / (float)LSEQ) : 1.f;
    float cdv  = coef * dv;
#pragma unroll
    for (int j = 0; j < HALFD; j += 4) {
      float4 kq = *(const float4*)&kv[j];
      row[j]     += cdv * kq.x;
      row[j + 1] += cdv * kq.y;
      row[j + 2] += cdv * kq.z;
      row[j + 3] += cdv * kq.w;
    }
  }

  // final readout with last timestep key
  __syncthreads();
  kv[tid] = kbase[(size_t)(LSEQ - 1) * 512 + tid];
  __syncthreads();
  float c = 0.f;
#pragma unroll
  for (int j = 0; j < HALFD; j += 4) {
    float4 kq = *(const float4*)&kv[j];
    c += row[j]     * kq.x;
    c += row[j + 1] * kq.y;
    c += row[j + 2] * kq.z;
    c += row[j + 3] * kq.w;
  }
  ws_c[b * 512 + which * HALFD + tid] = c;
}

// =====================================================================
// Kernel 3: logits = C (64x512) @ out_w (512xV) + out_b, f32 WMMA 16x16x4.
// Each wave owns one 16-col tile and all 4 M-tiles -> out_w streamed once.
// =====================================================================
__global__ __launch_bounds__(256) void outgemm_kernel(
    const float* __restrict__ C, const float* __restrict__ out_w,
    const float* __restrict__ out_b, float* __restrict__ out)
{
  const int tid  = threadIdx.x;
  const int lane = tid & 31, wv = tid >> 5;
  const int l15  = lane & 15, hf = lane >> 4;
  const int n0   = (blockIdx.x * 8 + wv) * 16;
  const int n    = n0 + l15;

  v8f acc[4] = {{}, {}, {}, {}};
  for (int k0 = 0; k0 < 512; k0 += 4) {
    int kb = k0 + 2 * hf;                   // f32 B 4x16: elem e -> K = e + 2*half
    v2f bfrag;
    bfrag.x = out_w[(size_t)kb * VOC + n];
    bfrag.y = out_w[(size_t)(kb + 1) * VOC + n];
    for (int mt = 0; mt < 4; ++mt) {
      v2f afrag = *(const v2f*)&C[(size_t)(mt * 16 + l15) * 512 + kb];
      acc[mt] = __builtin_amdgcn_wmma_f32_16x16x4_f32(false, afrag, false, bfrag,
                                                      (short)0, acc[mt], false, false);
    }
  }
  float bias = out_b[n];
  for (int mt = 0; mt < 4; ++mt)
    for (int r = 0; r < 8; ++r) {
      int m = mt * 16 + r + 8 * hf;
      out[(size_t)m * VOC + n] = acc[mt][r] + bias;
    }
}

// =====================================================================
extern "C" void kernel_launch(void* const* d_in, const int* in_sizes, int n_in,
                              void* d_out, int out_size, void* d_ws, size_t ws_size,
                              hipStream_t stream)
{
  const int*   seq   = (const int*)  d_in[0];
  const float* embed = (const float*)d_in[1];
  const float* ff_w1 = (const float*)d_in[2];
  const float* ff_b1 = (const float*)d_in[3];
  const float* ff_w2 = (const float*)d_in[4];
  const float* ff_b2 = (const float*)d_in[5];
  const float* ln_g  = (const float*)d_in[6];
  const float* ln_b  = (const float*)d_in[7];
  const float* sem_w = (const float*)d_in[8];
  const float* sem_b = (const float*)d_in[9];
  const float* epi_w = (const float*)d_in[10];
  const float* epi_b = (const float*)d_in[11];
  const float* out_w = (const float*)d_in[12];
  const float* out_b = (const float*)d_in[13];

  // workspace layout (total ~131 MB)
  char* p = (char*)d_ws;
  __bf16* w1t = (__bf16*)p;  p += (size_t)H2d * Hd * 2;          // 1 MB
  __bf16* w2t = (__bf16*)p;  p += (size_t)Hd * H2d * 2;          // 1 MB
  __bf16* pt  = (__bf16*)p;  p += (size_t)Hd * Hd * 2;           // 0.5 MB
  float*  wsk = (float*)p;   p += (size_t)TOK * 512 * 4;         // 128 MB
  float*  wsc = (float*)p;   p += (size_t)BSZ * 512 * 4;         // 128 KB

  prep_kernel<<<(H2d * Hd + Hd * H2d + Hd * Hd) / 256, 256, 0, stream>>>(
      ff_w1, ff_w2, sem_w, epi_w, w1t, w2t, pt);

  encoder_kernel<<<TOK / 16, 256, 0, stream>>>(
      seq, embed, w1t, ff_b1, w2t, ff_b2, ln_g, ln_b, pt, sem_b, epi_b, wsk);

  scan_kernel<<<BSZ * 2, 256, 0, stream>>>(wsk, wsc);

  outgemm_kernel<<<VOC / 128, 256, 0, stream>>>(wsc, out_w, out_b, (float*)d_out);
}
